// MLA_25443386262318
// MI455X (gfx1250) — matrix operand order repaired and checked by reading, not compiled, not run on hardware
//
#include <hip/hip_runtime.h>
#include <hip/hip_bf16.h>
#include <stdint.h>

// ---------------------------------------------------------------------------
// MLA prefill (DeepSeek-style) with lightning-indexer top-k sparse attention.
// All GEMM-shaped math goes through v_wmma_f32_16x16x32_bf16 (CDNA5 / gfx1250,
// wave32). f32 VALU for softmax / norms / RoPE / radix top-k selection.
// ---------------------------------------------------------------------------

typedef __bf16 bf16;
typedef __attribute__((ext_vector_type(16))) __bf16 v16bf;
typedef __attribute__((ext_vector_type(8)))  float  v8f;

constexpr int   kS    = 2048;   // sequence length
constexpr int   kDim  = 2048;   // model dim
constexpr int   kNH   = 16;     // heads
constexpr int   kNope = 128;    // QK_NOPE
constexpr int   kRope = 64;     // QK_ROPE
constexpr int   kQL   = 1536;   // Q_LORA
constexpr int   kKVL  = 512;    // KV_LORA
constexpr int   kIH   = 128;    // IDX_HD
constexpr int   kTopK = 1024;
constexpr float kNeg  = -1e30f;

union FragBF { v16bf v; uint4 q[2]; };

// A-operand (16x32, MxK): lane holds row M=lane&15; K split {k0..k0+7}+{k0+16..k0+23}
__device__ __forceinline__ v16bf loadA_frag(const bf16* row, int kbase, int lane) {
  FragBF f;
  const int k0 = kbase + ((lane & 16) ? 8 : 0);
  f.q[0] = *reinterpret_cast<const uint4*>(row + k0);
  f.q[1] = *reinterpret_cast<const uint4*>(row + k0 + 16);
  return f.v;
}
// B-operand (32x16, KxN): lane holds column N=lane&15; K split {k0..k0+7}+{k0+16..k0+23}
__device__ __forceinline__ v16bf loadB_frag(const bf16* row, int kbase, int lane) {
  FragBF f;
  const int k0 = kbase + ((lane & 16) ? 8 : 0);
  f.q[0] = *reinterpret_cast<const uint4*>(row + k0);
  f.q[1] = *reinterpret_cast<const uint4*>(row + k0 + 16);
  return f.v;
}
__device__ __forceinline__ v8f wmma_bf(v16bf a, v16bf b, v8f c) {
  return __builtin_amdgcn_wmma_f32_16x16x32_bf16(false, a, false, b, (short)0, c,
                                                 false, false);
}

// ----------------------------- generic batched GEMM ------------------------
// C[b][m][n] = sum_k A[b][m][k] * W[b][n][k]   (bf16 in, f32 accumulate)
template <bool BF_OUT>
__global__ __launch_bounds__(128) void k_gemm(
    const bf16* __restrict__ A, long lda, long sA,
    const bf16* __restrict__ W, long ldw, long sW,
    float* __restrict__ C, bf16* __restrict__ Cb, long ldc, long sC,
    int M, int N, int K)
{
  const int lane = threadIdx.x & 31;
  const int wave = threadIdx.x >> 5;
  const int tn = blockIdx.x * 4 + wave;
  const int tm = blockIdx.y;
  const int b  = blockIdx.z;
  if (tn * 16 >= N) return;  // wave-uniform
  const bf16* Ar = A + (size_t)b * sA + (size_t)(tm * 16 + (lane & 15)) * lda;
  const bf16* Wr = W + (size_t)b * sW + (size_t)(tn * 16 + (lane & 15)) * ldw;
  v8f acc = {};
  for (int k = 0; k < K; k += 32)
    acc = wmma_bf(loadA_frag(Ar, k, lane), loadB_frag(Wr, k, lane), acc);
  const int n  = tn * 16 + (lane & 15);
  const int mb = tm * 16 + ((lane & 16) ? 8 : 0);
#pragma unroll
  for (int r = 0; r < 8; ++r) {
    if constexpr (BF_OUT)
      Cb[(size_t)b * sC + (size_t)(mb + r) * ldc + n] = (bf16)acc[r];
    else
      C[(size_t)b * sC + (size_t)(mb + r) * ldc + n] = acc[r];
  }
}

// ----------------------------- small helpers -------------------------------
__global__ void k_cvt(const float* __restrict__ in, bf16* __restrict__ out, long n) {
  long i = (long)blockIdx.x * 256 + threadIdx.x;
  if (i < n) out[i] = (bf16)in[i];
}
__global__ void k_scale_cast(const float* __restrict__ in, bf16* __restrict__ out,
                             float sc, long n) {
  long i = (long)blockIdx.x * 256 + threadIdx.x;
  if (i < n) out[i] = (bf16)(in[i] * sc);
}
__global__ void k_hadamard(bf16* __restrict__ H) {  // Sylvester 128x128
  int i = blockIdx.x * 256 + threadIdx.x;
  int r = i >> 7, c = i & 127;
  H[i] = (bf16)((__popc(r & c) & 1) ? -1.0f : 1.0f);
}
// kv_bT[h][c][d] = kv_b[h*256+d][c]   (per-head transpose of first 128 rows)
__global__ void k_kvbT(const float* __restrict__ kvb, bf16* __restrict__ kvbT) {
  int i = blockIdx.x * 256 + threadIdx.x;   // over 16*512*128
  int d = i & 127, c = (i >> 7) & 511, h = i >> 16;
  kvbT[i] = (bf16)kvb[((size_t)(h * 256 + d)) * 512 + c];
}
__global__ void k_rms(const float* __restrict__ X, const float* __restrict__ w,
                      bf16* __restrict__ Y, int width) {
  int row = blockIdx.x, tid = threadIdx.x;
  __shared__ float red[256];
  const float* x = X + (size_t)row * width;
  float ss = 0.f;
  for (int i = tid; i < width; i += 256) { float v = x[i]; ss += v * v; }
  red[tid] = ss; __syncthreads();
  for (int o = 128; o > 0; o >>= 1) { if (tid < o) red[tid] += red[tid + o]; __syncthreads(); }
  float inv = rsqrtf(red[0] / (float)width + 1e-6f);
  for (int i = tid; i < width; i += 256)
    Y[(size_t)row * width + i] = (bf16)(x[i] * inv * w[i]);
}
// rms on kvp[:,0:512] -> kv & kvT; interleaved RoPE on kvp[:,512:576] -> k_pe
__global__ void k_kv_post(const float* __restrict__ kvp, const float* __restrict__ w,
                          const float* __restrict__ cosT, const float* __restrict__ sinT,
                          bf16* __restrict__ kv, bf16* __restrict__ kvT,
                          bf16* __restrict__ kpe) {
  int s = blockIdx.x, tid = threadIdx.x;
  __shared__ float red[256];
  const float* x = kvp + (size_t)s * 576;
  float ss = 0.f;
  for (int i = tid; i < kKVL; i += 256) { float v = x[i]; ss += v * v; }
  red[tid] = ss; __syncthreads();
  for (int o = 128; o > 0; o >>= 1) { if (tid < o) red[tid] += red[tid + o]; __syncthreads(); }
  float inv = rsqrtf(red[0] / (float)kKVL + 1e-6f);
  for (int i = tid; i < kKVL; i += 256) {
    bf16 v = (bf16)(x[i] * inv * w[i]);
    kv[(size_t)s * kKVL + i] = v;
    kvT[(size_t)i * kS + s] = v;
  }
  if (tid < 32) {
    float x1 = x[512 + 2 * tid], x2 = x[512 + 2 * tid + 1];
    float c = cosT[s * 32 + tid], sn = sinT[s * 32 + tid];
    kpe[(size_t)s * 64 + 2 * tid]     = (bf16)(x1 * c - x2 * sn);
    kpe[(size_t)s * 64 + 2 * tid + 1] = (bf16)(x1 * sn + x2 * c);
  }
}
// split q (S, NH, 192) -> q_nope bf16 + interleaved-RoPE q_pe bf16
__global__ void k_rope_q(const float* __restrict__ q, const float* __restrict__ cosT,
                         const float* __restrict__ sinT, bf16* __restrict__ qnope,
                         bf16* __restrict__ qpe) {
  int s = blockIdx.x, h = blockIdx.y, tid = threadIdx.x;
  const float* row = q + (size_t)s * (kNH * 192) + h * 192;
  qnope[((size_t)s * kNH + h) * kNope + tid] = (bf16)row[tid];
  if (tid < 32) {
    float x1 = row[128 + 2 * tid], x2 = row[128 + 2 * tid + 1];
    float c = cosT[s * 32 + tid], sn = sinT[s * 32 + tid];
    qpe[((size_t)s * kNH + h) * 64 + 2 * tid]     = (bf16)(x1 * c - x2 * sn);
    qpe[((size_t)s * kNH + h) * 64 + 2 * tid + 1] = (bf16)(x1 * sn + x2 * c);
  }
}
// half-RoPE on first 64 dims of iq (S,NH,128) -> bf16
__global__ void k_rope_iq(const float* __restrict__ iq, const float* __restrict__ cosT,
                          const float* __restrict__ sinT, bf16* __restrict__ out) {
  int s = blockIdx.x, h = blockIdx.y, d = threadIdx.x;
  const float* row = iq + (size_t)s * (kNH * kIH) + h * kIH;
  size_t o = ((size_t)s * kNH + h) * kIH;
  float v;
  if (d < 32)       v = row[d] * cosT[s * 32 + d] - row[d + 32] * sinT[s * 32 + d];
  else if (d < 64) { int j = d - 32; v = row[j] * sinT[s * 32 + j] + row[d] * cosT[s * 32 + j]; }
  else              v = row[d];
  out[o + d] = (bf16)v;
}
// layernorm(128) + half-RoPE on ik row -> bf16
__global__ void k_ik(const float* __restrict__ ikp, const float* __restrict__ w,
                     const float* __restrict__ b, const float* __restrict__ cosT,
                     const float* __restrict__ sinT, bf16* __restrict__ out) {
  int s = blockIdx.x, tid = threadIdx.x;
  __shared__ float red[128];
  __shared__ float ysh[128];
  const float* x = ikp + (size_t)s * kIH;
  float v = x[tid];
  red[tid] = v; __syncthreads();
  for (int o = 64; o > 0; o >>= 1) { if (tid < o) red[tid] += red[tid + o]; __syncthreads(); }
  float mu = red[0] / 128.f;
  __syncthreads();
  float dv = v - mu;
  red[tid] = dv * dv; __syncthreads();
  for (int o = 64; o > 0; o >>= 1) { if (tid < o) red[tid] += red[tid + o]; __syncthreads(); }
  float inv = rsqrtf(red[0] / 128.f + 1e-6f);
  float y = dv * inv * w[tid] + b[tid];
  ysh[tid] = y; __syncthreads();
  float r;
  if (tid < 32)       r = ysh[tid] * cosT[s * 32 + tid] - ysh[tid + 32] * sinT[s * 32 + tid];
  else if (tid < 64) { int j = tid - 32; r = ysh[j] * sinT[s * 32 + j] + ysh[tid] * cosT[s * 32 + j]; }
  else                r = y;
  out[(size_t)s * kIH + tid] = (bf16)r;
}
// wq = bf16( iw * 16^-.5 * (iq@H) * 128^-.5 * 128^-.5 )
__global__ void k_make_wq(const float* __restrict__ iqrot, const float* __restrict__ iw,
                          bf16* __restrict__ wq, long n) {
  long i = (long)blockIdx.x * 256 + threadIdx.x;
  if (i < n) wq[i] = (bf16)(iw[i >> 7] * iqrot[i] * 0.001953125f);  // 0.25/128
}

// index_score[s][t] = sum_h relu(wq[s,h]·ik[t]) for t<=s, else -1e30
__global__ __launch_bounds__(128) void k_iscore(const bf16* __restrict__ wq,
                                                const bf16* __restrict__ ik,
                                                float* __restrict__ iscore) {
  const int lane = threadIdx.x & 31, wave = threadIdx.x >> 5;
  const int tt = blockIdx.x * 4 + wave;
  const int st = blockIdx.y;
  if (tt * 16 > st * 16 + 15) return;  // fully above diagonal, never read
  const bf16* Ar = wq + (size_t)(st * 16 + (lane & 15)) * (kNH * kIH);
  const bf16* Wr = ik + (size_t)(tt * 16 + (lane & 15)) * kIH;
  float racc[8] = {0, 0, 0, 0, 0, 0, 0, 0};
  for (int h = 0; h < kNH; ++h) {
    v8f acc = {};
#pragma unroll
    for (int k = 0; k < kIH; k += 32)
      acc = wmma_bf(loadA_frag(Ar + h * kIH, k, lane), loadB_frag(Wr, k, lane), acc);
#pragma unroll
    for (int r = 0; r < 8; ++r) racc[r] += fmaxf(acc[r], 0.f);
  }
  const int t  = tt * 16 + (lane & 15);
  const int mb = st * 16 + ((lane & 16) ? 8 : 0);
#pragma unroll
  for (int r = 0; r < 8; ++r) {
    int s = mb + r;
    iscore[(size_t)s * kS + t] = (t <= s) ? racc[r] : kNeg;
  }
}

// exact top-1024 per row via bitwise radix select (scores >= 0 so uint order
// == float order), index-order tie-break to match jax.lax.top_k
__global__ __launch_bounds__(256) void k_select(const float* __restrict__ iscore,
                                                unsigned char* __restrict__ sel) {
  const int row = blockIdx.x, tid = threadIdx.x;
  __shared__ unsigned int bits[kS];
  __shared__ unsigned int red[256];
  __shared__ unsigned int runbase;
  const int valid = row + 1;
  for (int t = tid; t < valid; t += 256)
    bits[t] = __float_as_uint(iscore[(size_t)row * kS + t]);
  for (int t = valid + tid; t < kS; t += 256) sel[(size_t)row * kS + t] = 0;
  __syncthreads();
  if (valid <= kTopK) {
    for (int t = tid; t < valid; t += 256) sel[(size_t)row * kS + t] = 1;
    return;
  }
  unsigned int prefix = 0;
  int k = kTopK;
  for (int b = 31; b >= 0; --b) {
    unsigned int cand = prefix | (1u << b);
    unsigned int mhi  = ~((1u << b) - 1u);
    unsigned int c = 0;
    for (int t = tid; t < valid; t += 256)
      if ((bits[t] & mhi) == cand) ++c;
    red[tid] = c; __syncthreads();
    for (int o = 128; o > 0; o >>= 1) { if (tid < o) red[tid] += red[tid + o]; __syncthreads(); }
    unsigned int cnt = red[0];
    __syncthreads();
    if (cnt >= (unsigned)k) prefix = cand; else k -= (int)cnt;
  }
  const unsigned int thr = prefix;
  unsigned int cg = 0;
  for (int t = tid; t < valid; t += 256) if (bits[t] > thr) ++cg;
  red[tid] = cg; __syncthreads();
  for (int o = 128; o > 0; o >>= 1) { if (tid < o) red[tid] += red[tid + o]; __syncthreads(); }
  const int need_eq = kTopK - (int)red[0];
  __syncthreads();
  if (tid == 0) runbase = 0;
  __syncthreads();
  for (int base = 0; base < valid; base += 256) {
    int t = base + tid;
    unsigned int f = (t < valid && bits[t] == thr) ? 1u : 0u;
    red[tid] = f; __syncthreads();
    for (int o = 1; o < 256; o <<= 1) {
      unsigned int v = (tid >= o) ? red[tid - o] : 0u;
      __syncthreads();
      red[tid] += v;
      __syncthreads();
    }
    unsigned int excl = red[tid] - f;
    if (t < valid) {
      bool pick = (bits[t] > thr) || (f && (runbase + excl) < (unsigned)need_eq);
      sel[(size_t)row * kS + t] = pick ? 1 : 0;
    }
    unsigned int tot = red[255];
    __syncthreads();
    if (tid == 0) runbase += tot;
    __syncthreads();
  }
}

// Flash-style sparse attention. Block = (16-row s-tile, head), 8 waves.
// wave0: WMMA score tile + online softmax; all waves: P@kv^T WMMA on 64-col slice.
__global__ __launch_bounds__(256) void k_attn(
    const bf16* __restrict__ qabs, const bf16* __restrict__ qpe,
    const bf16* __restrict__ kv,   const bf16* __restrict__ kvT,
    const bf16* __restrict__ kpe,  const unsigned char* __restrict__ sel,
    bf16* __restrict__ omid)
{
  const int st = blockIdx.x;
  const int h  = blockIdx.y;
  const int lane = threadIdx.x & 31;
  const int wave = threadIdx.x >> 5;
  const int col  = lane & 15;
  const int mb   = (lane & 16) ? 8 : 0;

  __shared__ bf16  Pld[16 * 32];
  __shared__ float sold[16];
  __shared__ float srow[16];

  const float sm = 0.07216878364870322f;  // 192^-0.5

  float run_m[8], run_l[8];
#pragma unroll
  for (int r = 0; r < 8; ++r) { run_m[r] = -3.0e38f; run_l[r] = 0.f; }
  v8f acc[4];
  { v8f z = {};
#pragma unroll
    for (int i = 0; i < 4; ++i) acc[i] = z; }

  const int c0 = wave * 64;
  const int nchunk = (st * 16 + 15) / 32 + 1;

  for (int ch = 0; ch < nchunk; ++ch) {
    const int t0 = ch * 32;
    if (wave == 0) {
      const bf16* Ar  = qabs + (size_t)(st * 16 + col) * (kNH * kKVL) + h * kKVL;
      const bf16* Ar2 = qpe  + (size_t)(st * 16 + col) * (kNH * kRope) + h * kRope;
      float v0[8], v1[8];
#pragma unroll
      for (int sub = 0; sub < 2; ++sub) {
        const int tb = t0 + sub * 16;
        const bf16* Wr  = kv  + (size_t)(tb + col) * kKVL;
        const bf16* Wr2 = kpe + (size_t)(tb + col) * kRope;
        v8f sc = {};
        for (int k = 0; k < kKVL; k += 32)
          sc = wmma_bf(loadA_frag(Ar, k, lane), loadB_frag(Wr, k, lane), sc);
#pragma unroll
        for (int k = 0; k < kRope; k += 32)
          sc = wmma_bf(loadA_frag(Ar2, k, lane), loadB_frag(Wr2, k, lane), sc);
#pragma unroll
        for (int r = 0; r < 8; ++r) {
          const int sg = st * 16 + mb + r;
          const int t  = tb + col;
          bool ok = (t <= sg) && (sel[(size_t)sg * kS + t] != 0);
          float v = ok ? (sc[r] * sm) : kNeg;
          if (sub == 0) v0[r] = v; else v1[r] = v;
        }
      }
#pragma unroll
      for (int r = 0; r < 8; ++r) {
        float cm = fmaxf(v0[r], v1[r]);
#pragma unroll
        for (int o = 1; o < 16; o <<= 1) cm = fmaxf(cm, __shfl_xor(cm, o, 32));
        float nm = fmaxf(run_m[r], cm);
        float so = __expf(run_m[r] - nm);
        run_m[r] = nm;
        float p0 = __expf(v0[r] - nm);
        float p1 = __expf(v1[r] - nm);
        float ps = p0 + p1;
#pragma unroll
        for (int o = 1; o < 16; o <<= 1) ps += __shfl_xor(ps, o, 32);
        run_l[r] = run_l[r] * so + ps;
        Pld[(mb + r) * 32 + col]      = (bf16)p0;
        Pld[(mb + r) * 32 + 16 + col] = (bf16)p1;
        if (col == 0) sold[mb + r] = so;
      }
    }
    __syncthreads();
#pragma unroll
    for (int i = 0; i < 4; ++i)
#pragma unroll
      for (int r = 0; r < 8; ++r) acc[i][r] *= sold[mb + r];
    v16bf pf = loadA_frag(&Pld[col * 32], 0, lane);
#pragma unroll
    for (int i = 0; i < 4; ++i) {
      const bf16* Wr = kvT + (size_t)(c0 + i * 16 + col) * kS + t0;
      acc[i] = wmma_bf(pf, loadB_frag(Wr, 0, lane), acc[i]);
    }
    __syncthreads();
  }
  if (wave == 0 && col == 0) {
#pragma unroll
    for (int r = 0; r < 8; ++r) srow[mb + r] = run_l[r];
  }
  __syncthreads();
#pragma unroll
  for (int i = 0; i < 4; ++i)
#pragma unroll
    for (int r = 0; r < 8; ++r) {
      const int sg = st * 16 + mb + r;
      const int c  = c0 + i * 16 + col;
      omid[(size_t)sg * (kNH * kKVL) + h * kKVL + c] = (bf16)(acc[i][r] / srow[mb + r]);
    }
}

// ---------------------------------------------------------------------------
extern "C" void kernel_launch(void* const* d_in, const int* in_sizes, int n_in,
                              void* d_out, int out_size, void* d_ws, size_t ws_size,
                              hipStream_t stream)
{
  (void)in_sizes; (void)n_in; (void)out_size; (void)ws_size;
  const float* x       = (const float*)d_in[0];
  const float* cosT    = (const float*)d_in[2];
  const float* sinT    = (const float*)d_in[3];
  const float* q_a     = (const float*)d_in[5];
  const float* q_a_ln  = (const float*)d_in[6];
  const float* q_b     = (const float*)d_in[7];
  const float* kv_a    = (const float*)d_in[8];
  const float* kv_a_ln = (const float*)d_in[9];
  const float* kv_b    = (const float*)d_in[10];
  const float* o_proj  = (const float*)d_in[11];
  const float* iwqb    = (const float*)d_in[12];
  const float* iwk     = (const float*)d_in[13];
  const float* iknw    = (const float*)d_in[14];
  const float* iknb    = (const float*)d_in[15];
  const float* iwp     = (const float*)d_in[16];
  float* out = (float*)d_out;

  char* base = (char*)d_ws;
  size_t off = 0;
  auto alloc = [&](size_t bytes) -> char* {
    char* p = base + off;
    off += (bytes + 255) & ~(size_t)255;
    return p;
  };
  bf16*  x_bf   = (bf16*)alloc((size_t)kS * kDim * 2);
  bf16*  qa_bf  = (bf16*)alloc((size_t)kQL * kDim * 2);
  bf16*  qb_bf  = (bf16*)alloc((size_t)(kNH * 192) * kQL * 2);
  bf16*  kva_bf = (bf16*)alloc((size_t)(kKVL + kRope) * kDim * 2);
  bf16*  kvb_bf = (bf16*)alloc((size_t)(kNH * 256) * kKVL * 2);
  bf16*  kvbT   = (bf16*)alloc((size_t)kNH * kKVL * 128 * 2);
  bf16*  op_bf  = (bf16*)alloc((size_t)kDim * (kNH * 128) * 2);
  bf16*  iwq_bf = (bf16*)alloc((size_t)(kNH * 128) * kQL * 2);
  bf16*  iwk_bf = (bf16*)alloc((size_t)128 * kDim * 2);
  bf16*  iwp_bf = (bf16*)alloc((size_t)kNH * kDim * 2);
  bf16*  H_bf   = (bf16*)alloc((size_t)128 * 128 * 2);
  float* qr_f   = (float*)alloc((size_t)kS * kQL * 4);
  bf16*  qr_bf  = (bf16*)alloc((size_t)kS * kQL * 2);
  float* q_f    = (float*)alloc((size_t)kS * (kNH * 192) * 4);
  bf16*  qnope  = (bf16*)alloc((size_t)kS * kNH * 128 * 2);
  bf16*  qpe    = (bf16*)alloc((size_t)kS * kNH * 64 * 2);
  float* kvp_f  = (float*)alloc((size_t)kS * 576 * 4);
  bf16*  kv_bf  = (bf16*)alloc((size_t)kS * kKVL * 2);
  bf16*  kvT_bf = (bf16*)alloc((size_t)kKVL * kS * 2);
  bf16*  kpe_bf = (bf16*)alloc((size_t)kS * 64 * 2);
  float* iq_f   = (float*)alloc((size_t)kS * kNH * 128 * 4);
  bf16*  iq_bf  = (bf16*)alloc((size_t)kS * kNH * 128 * 2);
  bf16*  wq_bf  = (bf16*)alloc((size_t)kS * kNH * 128 * 2);
  float* ikp_f  = (float*)alloc((size_t)kS * 128 * 4);
  bf16*  ikpre  = (bf16*)alloc((size_t)kS * 128 * 2);
  float* ikrot  = (float*)alloc((size_t)kS * 128 * 4);
  bf16*  ik_bf  = (bf16*)alloc((size_t)kS * 128 * 2);
  float* iw_f   = (float*)alloc((size_t)kS * kNH * 4);
  unsigned char* sel = (unsigned char*)alloc((size_t)kS * kS);
  bf16*  qabs   = (bf16*)alloc((size_t)kS * kNH * kKVL * 2);
  bf16*  omid   = (bf16*)alloc((size_t)kS * kNH * kKVL * 2);
  bf16*  ocat   = (bf16*)alloc((size_t)kS * kNH * 128 * 2);
  float* iqrot  = iq_f;   // alias: iq_f dead after k_rope_iq
  float* iscore = q_f;    // alias: q_f (24MB) dead after k_rope_q

  auto cvt = [&](const float* src, bf16* dst, long n) {
    k_cvt<<<dim3((unsigned)((n + 255) / 256)), dim3(256), 0, stream>>>(src, dst, n);
  };
  auto gemmF = [&](const bf16* A, long lda, long sA, const bf16* W, long ldw, long sW,
                   float* C, long ldc, long sC, int M, int N, int K, int batch) {
    dim3 g((N + 63) / 64, M / 16, batch);
    k_gemm<false><<<g, dim3(128), 0, stream>>>(A, lda, sA, W, ldw, sW, C, nullptr, ldc, sC, M, N, K);
  };
  auto gemmB = [&](const bf16* A, long lda, long sA, const bf16* W, long ldw, long sW,
                   bf16* Cb, long ldc, long sC, int M, int N, int K, int batch) {
    dim3 g((N + 63) / 64, M / 16, batch);
    k_gemm<true><<<g, dim3(128), 0, stream>>>(A, lda, sA, W, ldw, sW, nullptr, Cb, ldc, sC, M, N, K);
  };

  // ---- bf16 staging ----
  cvt(x, x_bf, (long)kS * kDim);
  cvt(q_a, qa_bf, (long)kQL * kDim);
  cvt(q_b, qb_bf, (long)kNH * 192 * kQL);
  cvt(kv_a, kva_bf, (long)(kKVL + kRope) * kDim);
  cvt(kv_b, kvb_bf, (long)kNH * 256 * kKVL);
  cvt(o_proj, op_bf, (long)kDim * kNH * 128);
  cvt(iwqb, iwq_bf, (long)kNH * 128 * kQL);
  cvt(iwk, iwk_bf, (long)128 * kDim);
  cvt(iwp, iwp_bf, (long)kNH * kDim);
  k_kvbT<<<dim3(4096), dim3(256), 0, stream>>>(kv_b, kvbT);
  k_hadamard<<<dim3(64), dim3(256), 0, stream>>>(H_bf);

  // ---- projections + norms + rope ----
  gemmF(x_bf, kDim, 0, qa_bf, kDim, 0, qr_f, kQL, 0, kS, kQL, kDim, 1);
  k_rms<<<dim3(kS), dim3(256), 0, stream>>>(qr_f, q_a_ln, qr_bf, kQL);
  gemmF(x_bf, kDim, 0, kva_bf, kDim, 0, kvp_f, 576, 0, kS, 576, kDim, 1);
  k_kv_post<<<dim3(kS), dim3(256), 0, stream>>>(kvp_f, kv_a_ln, cosT, sinT, kv_bf, kvT_bf, kpe_bf);
  gemmF(qr_bf, kQL, 0, qb_bf, kQL, 0, q_f, kNH * 192, 0, kS, kNH * 192, kQL, 1);
  k_rope_q<<<dim3(kS, kNH), dim3(128), 0, stream>>>(q_f, cosT, sinT, qnope, qpe);

  // ---- indexer ----
  gemmF(qr_bf, kQL, 0, iwq_bf, kQL, 0, iq_f, kNH * 128, 0, kS, kNH * 128, kQL, 1);
  k_rope_iq<<<dim3(kS, kNH), dim3(128), 0, stream>>>(iq_f, cosT, sinT, iq_bf);
  gemmF(iq_bf, 128, 0, H_bf, 128, 0, iqrot, 128, 0, kS * kNH, 128, 128, 1);
  gemmF(x_bf, kDim, 0, iwk_bf, kDim, 0, ikp_f, 128, 0, kS, 128, kDim, 1);
  k_ik<<<dim3(kS), dim3(128), 0, stream>>>(ikp_f, iknw, iknb, cosT, sinT, ikpre);
  gemmF(ikpre, 128, 0, H_bf, 128, 0, ikrot, 128, 0, kS, 128, 128, 1);
  k_scale_cast<<<dim3((kS * 128 + 255) / 256), dim3(256), 0, stream>>>(
      ikrot, ik_bf, 0.08838834764831845f, (long)kS * 128);
  gemmF(x_bf, kDim, 0, iwp_bf, kDim, 0, iw_f, kNH, 0, kS, kNH, kDim, 1);
  k_make_wq<<<dim3((kS * kNH * 128 + 255) / 256), dim3(256), 0, stream>>>(
      iqrot, iw_f, wq_bf, (long)kS * kNH * 128);
  k_iscore<<<dim3(kS / 64, kS / 16), dim3(128), 0, stream>>>(wq_bf, ik_bf, iscore);
  k_select<<<dim3(kS), dim3(256), 0, stream>>>(iscore, sel);

  // ---- attention ----
  gemmB(qnope, kNH * 128, 128, kvbT, 128, (long)kKVL * 128,
        qabs, kNH * kKVL, kKVL, kS, kKVL, 128, kNH);
  k_attn<<<dim3(kS / 16, kNH), dim3(256), 0, stream>>>(qabs, qpe, kv_bf, kvT_bf, kpe_bf, sel, omid);
  gemmB(omid, kNH * kKVL, kKVL, kvb_bf + (size_t)128 * kKVL, kKVL, (long)256 * kKVL,
        ocat, kNH * 128, 128, kS, 128, kKVL, kNH);
  gemmF(ocat, kNH * 128, 0, op_bf, kNH * 128, 0, out, kDim, 0, kS, kDim, kNH * 128, 1);
}
